// CausalityAwareTemporalGNN_75299366633614
// MI455X (gfx1250) — matrix operand order repaired and checked by reading, not compiled, not run on hardware
//
#include <hip/hip_runtime.h>
#include <math.h>

// ---------------- problem constants (match reference) ----------------
#define N_NODES   4096
#define IN_F      28
#define HEADS     8
#define CH        256        // channels per head
#define HC        2048       // HEADS*CH
#define HID       256
#define OUTF      128
#define NEDGE     131072
#define NEDGE_SL  (NEDGE + N_NODES)   // with self loops
#define NEG_SLOPE 0.2f

// ---------------- CDNA5 WMMA types ----------------
typedef __attribute__((ext_vector_type(16))) __bf16 v16bf;
typedef __attribute__((ext_vector_type(8)))  float  v8f;

union Frag16 {
  v16bf v;
  unsigned short u[16];
  uint4 q[2];
};
union FragC {
  v8f v;
  float f[8];
};

__device__ inline v8f wmma_bf16(v16bf a, v16bf b, v8f c) {
  // D = A(16x32 bf16) * B(32x16 bf16) + C(16x16 f32)
  return __builtin_amdgcn_wmma_f32_16x16x32_bf16(false, a, false, b, (short)0, c,
                                                 false, false);
}

__device__ inline unsigned short f2bf(float f) {
  unsigned int u = __float_as_uint(f);
  u += 0x7FFFu + ((u >> 16) & 1u);            // round-to-nearest-even
  return (unsigned short)(u >> 16);
}
__device__ inline float bf2f(unsigned short b) {
  return __uint_as_float(((unsigned int)b) << 16);
}
__device__ inline float atomic_add_f32(float* p, float v) {
  return __hip_atomic_fetch_add(p, v, __ATOMIC_RELAXED, __HIP_MEMORY_SCOPE_AGENT);
}
// order-preserving float<->uint encode for atomicMax on floats
__device__ inline unsigned int fenc(float f) {
  unsigned int b = __float_as_uint(f);
  return (b & 0x80000000u) ? ~b : (b | 0x80000000u);
}
__device__ inline float fdec(unsigned int u) {
  unsigned int b = (u & 0x80000000u) ? (u & 0x7FFFFFFFu) : ~u;
  return __uint_as_float(b);
}
__device__ inline float sigmoidf(float x) { return 1.0f / (1.0f + __expf(-x)); }

// ---------------- small utility kernels ----------------
__global__ void k_fill_u32(unsigned int* p, unsigned int v, long long n) {
  long long i = (long long)blockIdx.x * blockDim.x + threadIdx.x;
  long long stride = (long long)gridDim.x * blockDim.x;
  for (; i < n; i += stride) p[i] = v;
}

__global__ void k_sum(const float* __restrict__ x, long long n, float* acc) {
  long long i = (long long)blockIdx.x * blockDim.x + threadIdx.x;
  long long stride = (long long)gridDim.x * blockDim.x;
  float s = 0.f;
  for (; i < n; i += stride) s += x[i];
  atomic_add_f32(acc, s);
}

__global__ void k_f2bf(const float* __restrict__ s, unsigned short* __restrict__ d,
                       long long n) {
  long long i = (long long)blockIdx.x * blockDim.x + threadIdx.x;
  long long stride = (long long)gridDim.x * blockDim.x;
  for (; i < n; i += stride) d[i] = f2bf(s[i]);
}

// xp = x @ W  (N x 28) @ (28 x 2048), thread per (n, j); lanes over j -> coalesced W reads
__global__ void k_xform(const float* __restrict__ x, const float* __restrict__ W,
                        float* __restrict__ xp) {
  int j = blockIdx.x * blockDim.x + threadIdx.x;   // 0..HC-1
  int n = blockIdx.y;
  float acc = 0.f;
  const float* xr = x + (long)n * IN_F;
#pragma unroll
  for (int k = 0; k < IN_F; ++k) acc += xr[k] * W[(long)k * HC + j];
  xp[(long)n * HC + j] = acc;
}

// a_s[n,h] = sum_c xp[n,h,c]*a_src[h,c];  a_d likewise
__global__ void k_coef(const float* __restrict__ xp, const float* __restrict__ asrc,
                       const float* __restrict__ adst, float* __restrict__ a_s,
                       float* __restrict__ a_d) {
  int t = blockIdx.x * blockDim.x + threadIdx.x;   // n*HEADS + h
  int n = t >> 3, h = t & 7;
  const float* row = xp + (long)n * HC + h * CH;
  const float* ws = asrc + h * CH;
  const float* wd = adst + h * CH;
  float s = 0.f, d = 0.f;
  for (int c = 0; c < CH; ++c) { s += row[c] * ws[c]; d += row[c] * wd[c]; }
  a_s[t] = s;
  a_d[t] = d;
}

// we_dot[h] = sum_c We[0, h*C+c] * a_edge[h,c]
__global__ void k_wedot(const float* __restrict__ We, const float* __restrict__ aed,
                        float* __restrict__ wd) {
  int h = threadIdx.x;
  if (h < HEADS) {
    float s = 0.f;
    for (int c = 0; c < CH; ++c) s += We[h * CH + c] * aed[h * CH + c];
    wd[h] = s;
  }
}

// pass A: alpha = leakyrelu(a_s[src]+a_d[dst]+ea*we_dot) ; atomic segment-max per dst
__global__ void k_edge_alpha(const int* __restrict__ src, const int* __restrict__ dst,
                             const float* __restrict__ ea, const float* __restrict__ a_s,
                             const float* __restrict__ a_d, const float* __restrict__ wd,
                             const float* __restrict__ meansum,
                             float* __restrict__ alpha, unsigned int* __restrict__ mx) {
  int e = blockIdx.x * blockDim.x + threadIdx.x;
  if (e >= NEDGE_SL) return;
  int s, d; float eav;
  if (e < NEDGE) { s = src[e]; d = dst[e]; eav = ea[e]; }
  else { s = d = e - NEDGE; eav = meansum[0] * (1.0f / (float)NEDGE); }
#pragma unroll
  for (int h = 0; h < HEADS; ++h) {
    float a = a_s[s * HEADS + h] + a_d[d * HEADS + h] + eav * wd[h];
    a = (a > 0.f) ? a : NEG_SLOPE * a;
    alpha[(long)e * HEADS + h] = a;
    atomicMax(&mx[d * HEADS + h], fenc(a));
  }
}

// pass B: ex = exp(alpha - mx[dst]); atomic segment-sum -> den
__global__ void k_edge_exp(const int* __restrict__ dst, float* __restrict__ alpha,
                           const unsigned int* __restrict__ mx, float* __restrict__ den) {
  int e = blockIdx.x * blockDim.x + threadIdx.x;
  if (e >= NEDGE_SL) return;
  int d = (e < NEDGE) ? dst[e] : (e - NEDGE);
#pragma unroll
  for (int h = 0; h < HEADS; ++h) {
    float ex = __expf(alpha[(long)e * HEADS + h] - fdec(mx[d * HEADS + h]));
    alpha[(long)e * HEADS + h] = ex;
    atomic_add_f32(&den[d * HEADS + h], ex);
  }
}

// pass C: out[dst] += xp[src] * w  (block per edge, 256 threads x 8 channels)
__global__ __launch_bounds__(256)
void k_edge_scatter(const int* __restrict__ src, const int* __restrict__ dst,
                    const float* __restrict__ ex, const float* __restrict__ den,
                    const float* __restrict__ xp, float* __restrict__ out) {
  int e = blockIdx.x;
  int s, d;
  if (e < NEDGE) { s = src[e]; d = dst[e]; }
  else { s = d = e - NEDGE; }
  __shared__ float w[HEADS];
  if (threadIdx.x < HEADS)
    w[threadIdx.x] = ex[(long)e * HEADS + threadIdx.x] /
                     (den[d * HEADS + threadIdx.x] + 1e-16f);
  __syncthreads();
  const float* xr = xp + (long)s * HC;
  float* orow = out + (long)d * HC;
  for (int j = threadIdx.x; j < HC; j += 256) {
    int h = j >> 8;   // /CH
    atomic_add_f32(&orow[j], xr[j] * w[h]);
  }
}

// temporal: h_t = gat + b_t  -> bf16
__global__ void k_bias_bf(const float* __restrict__ g, const float* __restrict__ b,
                          unsigned short* __restrict__ o) {
  long long i = (long long)blockIdx.x * blockDim.x + threadIdx.x;
  if (i < (long long)N_NODES * HC) o[i] = f2bf(g[i] + b[i % HC]);
}

// causal: h_c[n,c] = mean_h gat[n,h,c] + b_c[c]  -> f32 + bf16
__global__ void k_headmean(const float* __restrict__ g, const float* __restrict__ bc,
                           float* __restrict__ hf, unsigned short* __restrict__ hb) {
  int t = blockIdx.x * blockDim.x + threadIdx.x;   // n*CH + c
  int n = t >> 8, c = t & 255;
  float s = 0.f;
#pragma unroll
  for (int h = 0; h < HEADS; ++h) s += g[(long)n * HC + h * CH + c];
  float v = s * (1.0f / HEADS) + bc[c];
  hf[t] = v;
  hb[t] = f2bf(v);
}

// LSTM gates (h0=c0=0 so Whh term vanishes): gate order i,f,g,o
__global__ void k_lstm(const float* __restrict__ g, const float* __restrict__ bih,
                       const float* __restrict__ bhh, float* __restrict__ h) {
  int t = blockIdx.x * blockDim.x + threadIdx.x;   // n*HID + j
  int n = t >> 8, j = t & 255;
  const float* gr = g + (long)n * (4 * HID);
  float gi = gr[j] + bih[j] + bhh[j];
  float gg = gr[2 * HID + j] + bih[2 * HID + j] + bhh[2 * HID + j];
  float go = gr[3 * HID + j] + bih[3 * HID + j] + bhh[3 * HID + j];
  float c = sigmoidf(gi) * tanhf(gg);
  h[t] = sigmoidf(go) * tanhf(c);
}

// concat [h_temporal | h_causal] -> bf16 (N x 512)
__global__ void k_concat_bf(const float* __restrict__ a, const float* __restrict__ b,
                            unsigned short* __restrict__ o) {
  int t = blockIdx.x * blockDim.x + threadIdx.x;   // n*512 + j
  int n = t >> 9, j = t & 511;
  float v = (j < HID) ? a[(long)n * HID + j] : b[(long)n * HID + (j - HID)];
  o[t] = f2bf(v);
}

// ---------------- dense GEMM: C = A(MxK,bf16) * W(NxK,bf16)^T (+bias, relu) ---------------
// one wave -> 16x64 output tile: A fragment reused over 4 B tiles; K step 32.
__global__ __launch_bounds__(32)
void k_gemm_bf16(const unsigned short* __restrict__ A, const unsigned short* __restrict__ W,
                 const float* __restrict__ bias, float* __restrict__ C,
                 int M, int N, int K, int relu) {
  const int lane = threadIdx.x;
  const int laneLow = lane & 15;
  const int hiA = (lane & 16) ? 8 : 0;    // A fragment K-split
  const int hiB = (lane & 16) ? 16 : 0;   // unused; B loads split K by hiA like A
  const int mt = blockIdx.x;
  const int nt0 = blockIdx.y * 4;
  const long row = (long)mt * 16 + laneLow;

  FragC c[4];
#pragma unroll
  for (int t = 0; t < 4; ++t)
#pragma unroll
    for (int r = 0; r < 8; ++r) c[t].f[r] = 0.f;

  for (int kb = 0; kb < K; kb += 32) {
    Frag16 a;
    const unsigned short* ap = A + row * K + kb + hiA;
    a.q[0] = *(const uint4*)(ap);
    a.q[1] = *(const uint4*)(ap + 16);
#pragma unroll
    for (int t = 0; t < 4; ++t) {
      Frag16 b;
      const unsigned short* bp =
          W + ((long)(nt0 + t) * 16 + laneLow) * K + kb + hiA;
      b.q[0] = *(const uint4*)(bp);
      b.q[1] = *(const uint4*)(bp + 16);
      c[t].v = wmma_bf16(a.v, b.v, c[t].v);
    }
  }
#pragma unroll
  for (int t = 0; t < 4; ++t) {
    int col = (nt0 + t) * 16 + laneLow;
    float bb = bias ? bias[col] : 0.f;
#pragma unroll
    for (int r = 0; r < 8; ++r) {
      long m = (long)mt * 16 + r + hiA;
      float v = c[t].f[r] + bb;
      if (relu) v = fmaxf(v, 0.f);
      C[m * N + col] = v;
    }
  }
}

// ---------------- flash attention (4 heads, hd=64, seq=4096) ----------------
// one wave per (16-query tile, head); 32 keys per iteration; online softmax via LDS.
__global__ __launch_bounds__(32)
void k_flash(const unsigned short* __restrict__ qkv, float* __restrict__ out) {
  const int lane = threadIdx.x;
  const int laneLow = lane & 15;
  const int hiA = (lane & 16) ? 8 : 0;
  const int hiB = (lane & 16) ? 16 : 0;
  const int h = blockIdx.y;   // head
  const int qt = blockIdx.x;  // query tile

  __shared__ __align__(16) float s_tile[16][32];
  __shared__ __align__(16) unsigned short p_tile[16][32];
  __shared__ __align__(16) unsigned short vs[32][64];
  __shared__ float m_s[16], l_s[16], corr_s[16];

  // Q fragments for the two hd k-steps (0..31, 32..63), loaded once
  Frag16 q0, q1;
  {
    const unsigned short* qp = qkv + ((long)qt * 16 + laneLow) * 768 + h * 64;
    const unsigned short* p0 = qp + 0 + hiA;
    q0.q[0] = *(const uint4*)(p0);
    q0.q[1] = *(const uint4*)(p0 + 16);
    const unsigned short* p1 = qp + 32 + hiA;
    q1.q[0] = *(const uint4*)(p1);
    q1.q[1] = *(const uint4*)(p1 + 16);
  }

  FragC o[4];
#pragma unroll
  for (int t = 0; t < 4; ++t)
#pragma unroll
    for (int r = 0; r < 8; ++r) o[t].f[r] = 0.f;
  if (lane < 16) { m_s[lane] = -3.0e38f; l_s[lane] = 0.f; }
  __syncthreads();

  for (int kt = 0; kt < N_NODES; kt += 32) {
    // stage V tile (32 keys x 64) into LDS, coalesced 128b loads
    {
      const unsigned short* vp = qkv + (long)(kt + lane) * 768 + 512 + h * 64;
      uint4* dv = (uint4*)&vs[lane][0];
      const uint4* sv = (const uint4*)vp;
      dv[0] = sv[0]; dv[1] = sv[1]; dv[2] = sv[2]; dv[3] = sv[3];
    }
    // S = Q @ K^T for two 16-key subtiles
#pragma unroll
    for (int sub = 0; sub < 2; ++sub) {
      FragC s;
#pragma unroll
      for (int r = 0; r < 8; ++r) s.f[r] = 0.f;
      const unsigned short* kp =
          qkv + (long)(kt + sub * 16 + laneLow) * 768 + 256 + h * 64;
      Frag16 b0, b1;
      const unsigned short* b0p = kp + 0 + hiA;
      b0.q[0] = *(const uint4*)(b0p);
      b0.q[1] = *(const uint4*)(b0p + 16);
      const unsigned short* b1p = kp + 32 + hiA;
      b1.q[0] = *(const uint4*)(b1p);
      b1.q[1] = *(const uint4*)(b1p + 16);
      s.v = wmma_bf16(q0.v, b0.v, s.v);
      s.v = wmma_bf16(q1.v, b1.v, s.v);
#pragma unroll
      for (int r = 0; r < 8; ++r)
        s_tile[r + hiA][sub * 16 + laneLow] = s.f[r] * 0.125f;  // 1/sqrt(64)
    }
    __syncthreads();

    // online softmax row update (lanes 0..15 each own one query row)
    if (lane < 16) {
      float mold = m_s[lane];
      float mx = mold;
      for (int j = 0; j < 32; ++j) mx = fmaxf(mx, s_tile[lane][j]);
      float cr = __expf(mold - mx);
      float l = l_s[lane] * cr;
      for (int j = 0; j < 32; ++j) {
        float p = __expf(s_tile[lane][j] - mx);
        l += p;
        p_tile[lane][j] = f2bf(p);
      }
      m_s[lane] = mx; l_s[lane] = l; corr_s[lane] = cr;
    }
    __syncthreads();

    // rescale running O by correction
    float cr[8];
#pragma unroll
    for (int r = 0; r < 8; ++r) cr[r] = corr_s[r + hiA];
#pragma unroll
    for (int t = 0; t < 4; ++t)
#pragma unroll
      for (int r = 0; r < 8; ++r) o[t].f[r] *= cr[r];

    // A fragment of P (16q x 32k) from LDS
    Frag16 pa;
#pragma unroll
    for (int e = 0; e < 8; ++e) {
      pa.u[e]     = p_tile[laneLow][hiA + e];
      pa.u[8 + e] = p_tile[laneLow][16 + hiA + e];
    }
    // O += P @ V  (4 d-tiles)
#pragma unroll
    for (int t = 0; t < 4; ++t) {
      Frag16 vb;
#pragma unroll
      for (int e = 0; e < 16; ++e) vb.u[e] = vs[(e < 8 ? hiA + e : 8 + hiA + e)][t * 16 + laneLow];
      o[t].v = wmma_bf16(pa.v, vb.v, o[t].v);
    }
    __syncthreads();  // protect vs/s_tile before next iteration
  }

  float inv[8];
#pragma unroll
  for (int r = 0; r < 8; ++r) inv[r] = 1.0f / l_s[r + hiA];
#pragma unroll
  for (int t = 0; t < 4; ++t)
#pragma unroll
    for (int r = 0; r < 8; ++r)
      out[((long)qt * 16 + r + hiA) * HID + h * 64 + t * 16 + laneLow] =
          o[t].f[r] * inv[r];
}

// ---------------- host side ----------------
static inline size_t align256(size_t x) { return (x + 255) & ~(size_t)255; }

extern "C" void kernel_launch(void* const* d_in, const int* in_sizes, int n_in,
                              void* d_out, int out_size, void* d_ws, size_t ws_size,
                              hipStream_t stream) {
  (void)in_sizes; (void)n_in; (void)out_size; (void)ws_size;
  const float* x       = (const float*)d_in[0];
  const int*   ei_t    = (const int*)d_in[1];
  const int*   ei_c    = (const int*)d_in[2];
  const float* ea_t    = (const float*)d_in[3];
  const float* ea_c    = (const float*)d_in[4];
  const float* W_t     = (const float*)d_in[5];
  const float* asrc_t  = (const float*)d_in[6];
  const float* adst_t  = (const float*)d_in[7];
  const float* We_t    = (const float*)d_in[8];
  const float* aed_t   = (const float*)d_in[9];
  const float* b_t     = (const float*)d_in[10];
  const float* W_c     = (const float*)d_in[11];
  const float* asrc_c  = (const float*)d_in[12];
  const float* adst_c  = (const float*)d_in[13];
  const float* We_c    = (const float*)d_in[14];
  const float* aed_c   = (const float*)d_in[15];
  const float* b_c     = (const float*)d_in[16];
  const float* lstm_Wih= (const float*)d_in[17];
  const float* lstm_bih= (const float*)d_in[19];
  const float* lstm_bhh= (const float*)d_in[20];
  const float* min_w   = (const float*)d_in[21];
  const float* min_b   = (const float*)d_in[22];
  const float* mout_w  = (const float*)d_in[23];
  const float* mout_b  = (const float*)d_in[24];
  const float* fus_w   = (const float*)d_in[25];
  const float* fus_b   = (const float*)d_in[26];
  const float* out_w   = (const float*)d_in[27];
  const float* out_b   = (const float*)d_in[28];

  // d_out segments: out (N x 128) | h_temporal (N x 256) | h_causal (N x 256)
  float* out_final  = (float*)d_out;
  float* h_temporal = out_final + (long)N_NODES * OUTF;
  float* h_causal   = h_temporal + (long)N_NODES * HID;

  // workspace layout
  char* p = (char*)d_ws;
  auto take = [&](size_t bytes) { char* r = p; p += align256(bytes); return r; };
  float*          xp      = (float*)take((size_t)N_NODES * HC * 4);
  float*          gat     = (float*)take((size_t)N_NODES * HC * 4);
  float*          alpha   = (float*)take((size_t)NEDGE_SL * HEADS * 4);
  float*          a_s     = (float*)take((size_t)N_NODES * HEADS * 4);
  float*          a_d     = (float*)take((size_t)N_NODES * HEADS * 4);
  unsigned int*   mx      = (unsigned int*)take((size_t)N_NODES * HEADS * 4);
  float*          den     = (float*)take((size_t)N_NODES * HEADS * 4);
  float*          wedot   = (float*)take(HEADS * 4);
  float*          meansum = (float*)take(4);
  unsigned short* ht_bf   = (unsigned short*)take((size_t)N_NODES * HC * 2);
  unsigned short* wih_bf  = (unsigned short*)take((size_t)4 * HID * HC * 2);
  float*          gates   = (float*)take((size_t)N_NODES * 4 * HID * 4);
  float*          hc_f    = (float*)take((size_t)N_NODES * HID * 4);
  unsigned short* hc_bf   = (unsigned short*)take((size_t)N_NODES * HID * 2);
  float*          qkv_f   = (float*)take((size_t)N_NODES * 3 * HID * 4);
  unsigned short* qkv_bf  = (unsigned short*)take((size_t)N_NODES * 3 * HID * 2);
  float*          attn_f  = (float*)take((size_t)N_NODES * HID * 4);
  unsigned short* attn_bf = (unsigned short*)take((size_t)N_NODES * HID * 2);
  unsigned short* minw_bf = (unsigned short*)take((size_t)3 * HID * HID * 2);
  unsigned short* moutw_bf= (unsigned short*)take((size_t)HID * HID * 2);
  unsigned short* fusw_bf = (unsigned short*)take((size_t)HID * 2 * HID * 2);
  unsigned short* outw_bf = (unsigned short*)take((size_t)OUTF * HID * 2);
  unsigned short* fin_bf  = (unsigned short*)take((size_t)N_NODES * 2 * HID * 2);
  float*          hfus_f  = (float*)take((size_t)N_NODES * HID * 4);
  unsigned short* hfus_bf = (unsigned short*)take((size_t)N_NODES * HID * 2);

  // weight conversions to bf16
  k_f2bf<<<512, 256, 0, stream>>>(lstm_Wih, wih_bf, (long long)4 * HID * HC);
  k_f2bf<<<128, 256, 0, stream>>>(min_w,  minw_bf,  (long long)3 * HID * HID);
  k_f2bf<<<64, 256, 0, stream>>>(mout_w, moutw_bf, (long long)HID * HID);
  k_f2bf<<<64, 256, 0, stream>>>(fus_w,  fusw_bf,  (long long)HID * 2 * HID);
  k_f2bf<<<32, 256, 0, stream>>>(out_w,  outw_bf,  (long long)OUTF * HID);

  auto run_gat = [&](const int* ei, const float* ea, const float* W,
                     const float* asrc, const float* adst, const float* We,
                     const float* aed) {
    const int* src = ei;
    const int* dst = ei + NEDGE;
    k_fill_u32<<<64, 256, 0, stream>>>((unsigned int*)mx, 0u, (long long)N_NODES * HEADS);
    k_fill_u32<<<64, 256, 0, stream>>>((unsigned int*)den, 0u, (long long)N_NODES * HEADS);
    k_fill_u32<<<2048, 256, 0, stream>>>((unsigned int*)gat, 0u, (long long)N_NODES * HC);
    k_fill_u32<<<1, 32, 0, stream>>>((unsigned int*)meansum, 0u, 1);
    k_sum<<<256, 256, 0, stream>>>(ea, NEDGE, meansum);
    k_xform<<<dim3(HC / 256, N_NODES), 256, 0, stream>>>(x, W, xp);
    k_coef<<<(N_NODES * HEADS) / 256, 256, 0, stream>>>(xp, asrc, adst, a_s, a_d);
    k_wedot<<<1, 32, 0, stream>>>(We, aed, wedot);
    int eg = (NEDGE_SL + 255) / 256;
    k_edge_alpha<<<eg, 256, 0, stream>>>(src, dst, ea, a_s, a_d, wedot, meansum,
                                         alpha, mx);
    k_edge_exp<<<eg, 256, 0, stream>>>(dst, alpha, mx, den);
    k_edge_scatter<<<NEDGE_SL, 256, 0, stream>>>(src, dst, alpha, den, xp, gat);
  };

  // ---------- temporal branch: GAT(concat) -> 1-step LSTM ----------
  run_gat(ei_t, ea_t, W_t, asrc_t, adst_t, We_t, aed_t);
  k_bias_bf<<<(N_NODES * HC) / 256, 256, 0, stream>>>(gat, b_t, ht_bf);
  // gates = h_t @ Wih^T : (4096 x 2048) x (1024 x 2048)
  k_gemm_bf16<<<dim3(N_NODES / 16, (4 * HID) / 64), 32, 0, stream>>>(
      ht_bf, wih_bf, nullptr, gates, N_NODES, 4 * HID, HC, 0);
  k_lstm<<<(N_NODES * HID) / 256, 256, 0, stream>>>(gates, lstm_bih, lstm_bhh,
                                                    h_temporal);

  // ---------- causal branch: GAT(mean) -> MHA ----------
  run_gat(ei_c, ea_c, W_c, asrc_c, adst_c, We_c, aed_c);
  k_headmean<<<(N_NODES * HID) / 256, 256, 0, stream>>>(gat, b_c, hc_f, hc_bf);
  // qkv = h_c @ in_w^T + in_b : (4096 x 256) x (768 x 256)
  k_gemm_bf16<<<dim3(N_NODES / 16, (3 * HID) / 64), 32, 0, stream>>>(
      hc_bf, minw_bf, min_b, qkv_f, N_NODES, 3 * HID, HID, 0);
  k_f2bf<<<2048, 256, 0, stream>>>(qkv_f, qkv_bf, (long long)N_NODES * 3 * HID);
  k_flash<<<dim3(N_NODES / 16, 4), 32, 0, stream>>>(qkv_bf, attn_f);
  k_f2bf<<<1024, 256, 0, stream>>>(attn_f, attn_bf, (long long)N_NODES * HID);
  // h_causal = attn @ out_w^T + out_b
  k_gemm_bf16<<<dim3(N_NODES / 16, HID / 64), 32, 0, stream>>>(
      attn_bf, moutw_bf, mout_b, h_causal, N_NODES, HID, HID, 0);

  // ---------- fusion + projection ----------
  k_concat_bf<<<(N_NODES * 2 * HID) / 256, 256, 0, stream>>>(h_temporal, h_causal,
                                                             fin_bf);
  k_gemm_bf16<<<dim3(N_NODES / 16, HID / 64), 32, 0, stream>>>(
      fin_bf, fusw_bf, fus_b, hfus_f, N_NODES, HID, 2 * HID, 1 /*relu*/);
  k_f2bf<<<1024, 256, 0, stream>>>(hfus_f, hfus_bf, (long long)N_NODES * HID);
  k_gemm_bf16<<<dim3(N_NODES / 16, OUTF / 64), 32, 0, stream>>>(
      hfus_bf, outw_bf, out_b, out_final, N_NODES, OUTF, HID, 0);
}
